// BertEncoder_80066780332619
// MI455X (gfx1250) — matrix operand order repaired and checked by reading, not compiled, not run on hardware
//
#include <hip/hip_runtime.h>
#include <hip/hip_bf16.h>

typedef __attribute__((ext_vector_type(16))) _Float16 v16h;
typedef __attribute__((ext_vector_type(8)))  float    v8f;
typedef __attribute__((ext_vector_type(4)))  unsigned int v4u;
typedef __attribute__((ext_vector_type(8)))  int v8i;
typedef __attribute__((ext_vector_type(4)))  int v4i;

#define NG   64      // num_graphs (reference B)
#define SMAX 512     // MAXN
#define HD   256     // hidden
#define LAYERS 4

#if __has_builtin(__builtin_amdgcn_tensor_load_to_lds) && __has_builtin(__builtin_amdgcn_s_wait_tensorcnt)
#define HAVE_TDM 1
#else
#define HAVE_TDM 0
#endif

// ---------------- WMMA helpers (CDNA5 wave32 layouts, ISA 7.12.2) ------------

__device__ __forceinline__ v8f wmma_f16(v16h a, v16h b, v8f c) {
  return __builtin_amdgcn_wmma_f32_16x16x32_f16(false, a, false, b, (short)0, c,
                                                false, false);
}

// A fragment: 16x32 f16, row-major source with leading dim lda (elements).
__device__ __forceinline__ v16h load_frag_a(const _Float16* src, int lda) {
  int lane = threadIdx.x & 31;
  int row  = lane & 15;
  int kofs = (lane >> 4) << 4;              // 0 or 16
  const _Float16* r = src + (size_t)row * lda + kofs;
  v16h a;
#pragma unroll
  for (int p = 0; p < 8; ++p) {
    int k = p << 1;                          // 0,2,4,...,14
    a[2 * p]     = r[k];
    a[2 * p + 1] = r[k + 1];
  }
  return a;
}

// B fragment: 32x16 from row-major f16 [K x Nfull] at (k0, col0).
__device__ __forceinline__ v16h load_frag_b_f16(const _Float16* src, int ldb) {
  int lane = threadIdx.x & 31;
  int col  = lane & 15;
  int kofs = (lane >> 4) << 4;              // 0 or 16
  const _Float16* c = src + (size_t)kofs * ldb + col;
  v16h b;
#pragma unroll
  for (int e = 0; e < 16; ++e) b[e] = c[(size_t)e * ldb];
  return b;
}

// B fragment for K^T in attention: contiguous dk -> one 32B vector per lane.
__device__ __forceinline__ v16h load_frag_b_ktrans(const _Float16* src, int ld) {
  int lane = threadIdx.x & 31;
  int key  = lane & 15;
  int kofs = (lane >> 4) << 4;
  return *(const v16h*)(src + (size_t)key * ld + kofs);
}

#if HAVE_TDM
// Tensor Data Mover: DMA a rows x cols f16 tile (row stride srcld elements)
// from global into LDS at byte offset lds_off.  D# per CDNA5 ISA ch.8.
__device__ __forceinline__ void tdm_load_tile_2d(unsigned lds_off,
                                                 const _Float16* gsrc,
                                                 int rows, int cols, int srcld) {
  unsigned long long ga = (unsigned long long)(uintptr_t)gsrc;
  v4u g0;
  g0[0] = 1u;                                      // count=1, user mode
  g0[1] = lds_off;                                 // lds_addr
  g0[2] = (unsigned)ga;                            // global_addr[31:0]
  g0[3] = (unsigned)((ga >> 32) & 0x01FFFFFFu) | (2u << 30);  // addr[56:32]|type=2
  v8i g1;
  g1[0] = 1 << 16;                                 // data_size = 2 bytes
  g1[1] = (int)((unsigned)(srcld & 0xFFFF) << 16); // tensor_dim0[15:0]
  g1[2] = (int)(((unsigned)(rows & 0xFFFF) << 16) |
                ((unsigned)srcld >> 16));          // tensor_dim1 lo | dim0 hi
  g1[3] = (int)((unsigned)(cols & 0xFFFF) << 16);  // tile_dim0 | dim1 hi
  g1[4] = rows & 0xFFFF;                           // tile_dim1 (tile_dim2=0)
  g1[5] = srcld;                                   // tensor_dim0_stride lo32
  g1[6] = 0;
  g1[7] = 0;
  v4i z4 = {};
  v8i z8 = {};
  __builtin_amdgcn_tensor_load_to_lds(g0, g1, z4, z4, z8, 0);
}
#endif

// Fallback / generic cooperative stage of a 32x128 f16 tile (256 threads).
__device__ __forceinline__ void stage_b_tile(_Float16 (*sB)[128],
                                             const _Float16* src, int ld) {
  int t = threadIdx.x;
  int r = t >> 3, cs = (t & 7) * 16;
  *(v16h*)&sB[r][cs] = *(const v16h*)(src + (size_t)r * ld + cs);
}

// Row LayerNorm over a [16][256] f32 LDS tile; blockDim must be 512.
__device__ __forceinline__ void block_ln(float (*sIn)[HD],
                                         const float* g, const float* b,
                                         _Float16* dsth, int ldh,
                                         float* dstf, int ldf, bool do_relu) {
  int t = threadIdx.x;
  int row = t >> 5, lane = t & 31;
  float vals[8];
  float s = 0.f;
#pragma unroll
  for (int j = 0; j < 8; ++j) { vals[j] = sIn[row][lane + 32 * j]; s += vals[j]; }
#pragma unroll
  for (int o = 16; o; o >>= 1) s += __shfl_xor(s, o, 32);
  float mean = s * (1.f / 256.f);
  float vsum = 0.f;
#pragma unroll
  for (int j = 0; j < 8; ++j) { float d = vals[j] - mean; vsum += d * d; }
#pragma unroll
  for (int o = 16; o; o >>= 1) vsum += __shfl_xor(vsum, o, 32);
  float rstd = rsqrtf(vsum * (1.f / 256.f) + 1e-5f);
#pragma unroll
  for (int j = 0; j < 8; ++j) {
    int c = lane + 32 * j;
    float y = (vals[j] - mean) * rstd * g[c] + b[c];
    if (do_relu) y = fmaxf(y, 0.f);
    if (dsth) dsth[(size_t)row * ldh + c] = (_Float16)y;
    if (dstf) dstf[(size_t)row * ldf + c] = y;
  }
}

// ---------------------------- setup kernels ---------------------------------

__global__ void k_starts(const int* __restrict__ batch, int n, int ng,
                         int* counts, int* starts) {
  __shared__ int c[256];
  int t = threadIdx.x;
  c[t] = 0;
  __syncthreads();
  for (int i = t; i < n; i += blockDim.x) atomicAdd(&c[batch[i]], 1);
  __syncthreads();
  if (t == 0) {
    int acc = 0;
    for (int g = 0; g < ng; ++g) { starts[g] = acc; counts[g] = c[g]; acc += c[g]; }
  }
}

__global__ void k_zero(float* p, _Float16* ph, size_t nelem) {
  size_t stride = (size_t)gridDim.x * blockDim.x;
  for (size_t i = (size_t)blockIdx.x * blockDim.x + threadIdx.x; i < nelem; i += stride) {
    p[i] = 0.f;
    ph[i] = (_Float16)0.f;
  }
}

__global__ void k_cvt(const float* __restrict__ s, _Float16* __restrict__ d, size_t n) {
  size_t stride = (size_t)gridDim.x * blockDim.x;
  for (size_t i = (size_t)blockIdx.x * blockDim.x + threadIdx.x; i < n; i += stride)
    d[i] = (_Float16)s[i];
}

// --------------------------- fused embedding --------------------------------
__global__ void __launch_bounds__(512)
k_embed(const float* __restrict__ x, const int* __restrict__ wl,
        const int* __restrict__ pos, const int* __restrict__ hop,
        const int* __restrict__ batch, const int* __restrict__ starts, int n,
        const _Float16* __restrict__ W_in, const float* __restrict__ b_in,
        const float* __restrict__ g_in, const float* __restrict__ be_in,
        const _Float16* __restrict__ W_fe, const float* __restrict__ b_fe,
        const float* __restrict__ g_fe, const float* __restrict__ be_fe,
        const float* __restrict__ E_wl, const float* __restrict__ E_pos,
        const float* __restrict__ E_hop,
        const _Float16* __restrict__ W_np, const float* __restrict__ b_np,
        float* __restrict__ p, _Float16* __restrict__ ph) {
  __shared__ _Float16 sA[16][1024];
  __shared__ float sOut[16][HD];
  int node0 = blockIdx.x * 16;
  int t = threadIdx.x, wave = t >> 5, lane = t & 31;
  int colw = wave * 16 + (lane & 15);
  int rbase = (lane >> 4) << 3;

  for (int i = t; i < 16 * 128; i += 512) {
    int r = i >> 7, c = i & 127;
    int node = node0 + r;
    sA[r][c] = (_Float16)((node < n) ? x[(size_t)node * 128 + c] : 0.f);
  }
  __syncthreads();

  {  // GEMM1: [16x128] @ W_in[128x256]
    v8f acc = {};
    for (int k0 = 0; k0 < 128; k0 += 32) {
      v16h a = load_frag_a(&sA[0][k0], 1024);
      v16h b = load_frag_b_f16(W_in + (size_t)k0 * HD + wave * 16, HD);
      acc = wmma_f16(a, b, acc);
    }
#pragma unroll
    for (int v = 0; v < 8; ++v) sOut[rbase + v][colw] = acc[v] + b_in[colw];
  }
  __syncthreads();
  block_ln(sOut, g_in, be_in, &sA[0][0], 1024, nullptr, 0, true);
  __syncthreads();

  {  // GEMM2: [16x256] @ W_fe[256x256]
    v8f acc = {};
    for (int k0 = 0; k0 < 256; k0 += 32) {
      v16h a = load_frag_a(&sA[0][k0], 1024);
      v16h b = load_frag_b_f16(W_fe + (size_t)k0 * HD + wave * 16, HD);
      acc = wmma_f16(a, b, acc);
    }
#pragma unroll
    for (int v = 0; v < 8; ++v) sOut[rbase + v][colw] = acc[v] + b_fe[colw];
  }
  __syncthreads();
  block_ln(sOut, g_fe, be_fe, &sA[0][0], 1024, nullptr, 0, true);
  for (int i = t; i < 16 * 768; i += 512) {
    int r = i / 768, c = i % 768;
    int node = node0 + r;
    float v = 0.f;
    if (node < n) {
      if (c < 256)      v = E_wl[(size_t)wl[node] * HD + c];
      else if (c < 512) v = E_pos[(size_t)pos[node] * HD + (c - 256)];
      else              v = E_hop[(size_t)hop[node] * HD + (c - 512)];
    }
    sA[r][256 + c] = (_Float16)v;
  }
  __syncthreads();

  {  // GEMM3: [16x1024] @ W_np[1024x256] -> scatter
    v8f acc = {};
    for (int k0 = 0; k0 < 1024; k0 += 32) {
      v16h a = load_frag_a(&sA[0][k0], 1024);
      v16h b = load_frag_b_f16(W_np + (size_t)k0 * HD + wave * 16, HD);
      acc = wmma_f16(a, b, acc);
    }
    float bb = b_np[colw];
#pragma unroll
    for (int v = 0; v < 8; ++v) {
      int node = node0 + rbase + v;
      if (node < n) {
        int g = batch[node];
        int slot = node - starts[g];
        size_t idx = ((size_t)g * SMAX + slot) * HD + colw;
        float val = acc[v] + bb;
        p[idx] = val;
        ph[idx] = (_Float16)val;
      }
    }
  }
}

// -------------- QKV GEMM: 64 rows x 128 cols, TDM-staged B tiles ------------
__global__ void __launch_bounds__(256)
k_qkv(const _Float16* __restrict__ phalf, const _Float16* __restrict__ Wq,
      const float* __restrict__ bq, _Float16* __restrict__ qkv) {
  __shared__ _Float16 sB[2][32][128];
  int t = threadIdx.x, wave = t >> 5, lane = t & 31;
  size_t row0 = (size_t)blockIdx.x * 64;
  int colbase = blockIdx.y * 128;
  v8f acc[4] = {{}, {}, {}, {}};

#if HAVE_TDM
  if (wave == 0)
    tdm_load_tile_2d((unsigned)(uintptr_t)&sB[0][0][0], Wq + colbase, 32, 128, 768);
#else
  stage_b_tile(sB[0], Wq + colbase, 768);
#endif
  for (int ks = 0; ks < 8; ++ks) {
    int buf = ks & 1;
#if HAVE_TDM
    if (wave == 0) {
      if (ks + 1 < 8) {
        tdm_load_tile_2d((unsigned)(uintptr_t)&sB[buf ^ 1][0][0],
                         Wq + (size_t)(ks + 1) * 32 * 768 + colbase, 32, 128, 768);
        __builtin_amdgcn_s_wait_tensorcnt(1);
      } else {
        __builtin_amdgcn_s_wait_tensorcnt(0);
      }
    }
#endif
    __syncthreads();
    v16h b = load_frag_b_f16(&sB[buf][0][wave * 16], 128);
#pragma unroll
    for (int r = 0; r < 4; ++r) {
      v16h a = load_frag_a(phalf + (row0 + r * 16) * HD + ks * 32, HD);
      acc[r] = wmma_f16(a, b, acc[r]);
    }
    __syncthreads();
#if !HAVE_TDM
    if (ks + 1 < 8)
      stage_b_tile(sB[buf ^ 1], Wq + (size_t)(ks + 1) * 32 * 768 + colbase, 768);
#endif
  }
  int col = colbase + wave * 16 + (lane & 15);
  int rbase = (lane >> 4) << 3;
  float bias = bq[col];
#pragma unroll
  for (int r = 0; r < 4; ++r)
#pragma unroll
    for (int v = 0; v < 8; ++v)
      qkv[(row0 + r * 16 + rbase + v) * 768 + col] = (_Float16)(acc[r][v] + bias);
}

// ------------------------- flash attention (WMMA) ---------------------------
__global__ void __launch_bounds__(256)
k_attn(const _Float16* __restrict__ qkv, const int* __restrict__ counts,
       _Float16* __restrict__ attno) {
  __shared__ _Float16 sP[8][16][32];
  int qt = blockIdx.x, g = blockIdx.y;
  int h = threadIdx.x >> 5, lane = threadIdx.x & 31;
  int cnt = counts[g];
  size_t base = (size_t)g * SMAX;
  int kl = lane & 15;
  int rbase = (lane >> 4) << 3;

  v16h qf = load_frag_a(qkv + (base + qt * 16) * 768 + h * 32, 768);
  v8f o0 = {}, o1 = {};
  float m[8], lsum[8];
#pragma unroll
  for (int v = 0; v < 8; ++v) { m[v] = -1e30f; lsum[v] = 0.f; }
  const float scale = 0.1767766952966369f;  // 1/sqrt(32)

  for (int kt = 0; kt < SMAX; kt += 32) {
    const _Float16* Kb = qkv + (base + kt) * 768 + 256 + h * 32;
    v16h bk0 = load_frag_b_ktrans(Kb, 768);
    v16h bk1 = load_frag_b_ktrans(Kb + (size_t)16 * 768, 768);
    v8f s0 = {}, s1 = {};
    s0 = wmma_f16(qf, bk0, s0);
    s1 = wmma_f16(qf, bk1, s1);
    float bias0 = ((kt + kl) < cnt) ? 0.f : -1e9f;
    float bias1 = ((kt + 16 + kl) < cnt) ? 0.f : -1e9f;
    float p0[8], p1[8];
#pragma unroll
    for (int v = 0; v < 8; ++v) {
      float a0 = s0[v] * scale + bias0;
      float a1 = s1[v] * scale + bias1;
      float tmx = fmaxf(a0, a1);
      tmx = fmaxf(tmx, __shfl_xor(tmx, 1, 32));
      tmx = fmaxf(tmx, __shfl_xor(tmx, 2, 32));
      tmx = fmaxf(tmx, __shfl_xor(tmx, 4, 32));
      tmx = fmaxf(tmx, __shfl_xor(tmx, 8, 32));
      float mnew = fmaxf(m[v], tmx);
      float alpha = __expf(m[v] - mnew);
      m[v] = mnew;
      p0[v] = __expf(a0 - mnew);
      p1[v] = __expf(a1 - mnew);
      float rs = p0[v] + p1[v];
      rs += __shfl_xor(rs, 1, 32);
      rs += __shfl_xor(rs, 2, 32);
      rs += __shfl_xor(rs, 4, 32);
      rs += __shfl_xor(rs, 8, 32);
      lsum[v] = lsum[v] * alpha + rs;
      o0[v] *= alpha;
      o1[v] *= alpha;
    }
    __syncthreads();
#pragma unroll
    for (int v = 0; v < 8; ++v) {
      sP[h][rbase + v][kl]      = (_Float16)p0[v];
      sP[h][rbase + v][16 + kl] = (_Float16)p1[v];
    }
    __syncthreads();
    v16h pa = load_frag_a(&sP[h][0][0], 32);
    const _Float16* Vb = qkv + (base + kt) * 768 + 512 + h * 32;
    v16h bv0 = load_frag_b_f16(Vb, 768);
    v16h bv1 = load_frag_b_f16(Vb + 16, 768);
    o0 = wmma_f16(pa, bv0, o0);
    o1 = wmma_f16(pa, bv1, o1);
  }
#pragma unroll
  for (int v = 0; v < 8; ++v) {
    float r = (lsum[v] > 0.f) ? 1.f / lsum[v] : 0.f;
    size_t row = base + qt * 16 + rbase + v;
    attno[row * HD + h * 32 + kl]      = (_Float16)(o0[v] * r);
    attno[row * HD + h * 32 + 16 + kl] = (_Float16)(o1[v] * r);
  }
}

// --------------- o-proj + LN1 + FFN(gelu) + LN2, fully fused ----------------
__global__ void __launch_bounds__(512)
k_otail(const _Float16* __restrict__ attno, const _Float16* __restrict__ Wo,
        const float* __restrict__ bo, const float* __restrict__ ln1g,
        const float* __restrict__ ln1b, const _Float16* __restrict__ W1,
        const float* __restrict__ b1, const _Float16* __restrict__ W2,
        const float* __restrict__ b2, const float* __restrict__ ln2g,
        const float* __restrict__ ln2b, float* __restrict__ p,
        _Float16* __restrict__ ph) {
  __shared__ float    sOut[16][HD];
  __shared__ _Float16 sH[16][HD];
  __shared__ _Float16 sFF[16][512];
  __shared__ float    sOut2[16][HD];
  size_t row0 = (size_t)blockIdx.x * 16;
  int t = threadIdx.x, wave = t >> 5, lane = t & 31;
  int colw = wave * 16 + (lane & 15);
  int rbase = (lane >> 4) << 3;

  {  // o @ Wo + bo + residual
    v8f acc = {};
    for (int k0 = 0; k0 < 256; k0 += 32) {
      v16h a = load_frag_a(attno + row0 * HD + k0, HD);
      v16h b = load_frag_b_f16(Wo + (size_t)k0 * HD + wave * 16, HD);
      acc = wmma_f16(a, b, acc);
    }
    float bb = bo[colw];
#pragma unroll
    for (int v = 0; v < 8; ++v) {
      int r = rbase + v;
      sOut[r][colw] = acc[v] + bb + p[(row0 + r) * HD + colw];
    }
  }
  __syncthreads();
  block_ln(sOut, ln1g, ln1b, &sH[0][0], HD, &sOut[0][0], HD, false);
  __syncthreads();

  v8f acc2 = {};
  for (int half = 0; half < 2; ++half) {
    int ffbase = half * 512;
#pragma unroll
    for (int j = 0; j < 2; ++j) {
      int fcol = ffbase + j * 256 + wave * 16;
      v8f a1 = {};
      for (int k0 = 0; k0 < 256; k0 += 32) {
        v16h a = load_frag_a(&sH[0][k0], HD);
        v16h b = load_frag_b_f16(W1 + (size_t)k0 * 1024 + fcol, 1024);
        a1 = wmma_f16(a, b, a1);
      }
      int lcol = j * 256 + wave * 16 + (lane & 15);
      float bb = b1[fcol + (lane & 15)];
#pragma unroll
      for (int v = 0; v < 8; ++v) {
        float xg = a1[v] + bb;
        xg = 0.5f * xg * (1.f + erff(xg * 0.7071067811865476f));  // exact gelu
        sFF[rbase + v][lcol] = (_Float16)xg;
      }
    }
    __syncthreads();
    for (int k0 = 0; k0 < 512; k0 += 32) {
      v16h a = load_frag_a(&sFF[0][k0], 512);
      v16h b = load_frag_b_f16(W2 + (size_t)(ffbase + k0) * HD + wave * 16, HD);
      acc2 = wmma_f16(a, b, acc2);
    }
    __syncthreads();
  }
  {
    float bb = b2[colw];
#pragma unroll
    for (int v = 0; v < 8; ++v) {
      int r = rbase + v;
      sOut2[r][colw] = acc2[v] + bb + sOut[r][colw];
    }
  }
  __syncthreads();
  block_ln(sOut2, ln2g, ln2b, ph + row0 * HD, HD, p + row0 * HD, HD, false);
}

// ------------------------------ gather --------------------------------------
__global__ void k_gather(const float* __restrict__ p, const int* __restrict__ batch,
                         const int* __restrict__ starts, int n,
                         float* __restrict__ out) {
  size_t total = (size_t)n * HD;
  size_t stride = (size_t)gridDim.x * blockDim.x;
  for (size_t i = (size_t)blockIdx.x * blockDim.x + threadIdx.x; i < total; i += stride) {
    int node = (int)(i >> 8);
    int c = (int)(i & 255);
    int g = batch[node];
    int slot = node - starts[g];
    out[i] = p[((size_t)g * SMAX + slot) * HD + c];
  }
}

// ------------------------------ launcher ------------------------------------
extern "C" void kernel_launch(void* const* d_in, const int* in_sizes, int n_in,
                              void* d_out, int out_size, void* d_ws, size_t ws_size,
                              hipStream_t stream) {
  const float* x     = (const float*)d_in[0];
  const int*   wl    = (const int*)d_in[1];
  const int*   pos   = (const int*)d_in[2];
  const int*   hop   = (const int*)d_in[3];
  const int*   batch = (const int*)d_in[4];
  const float* W_in  = (const float*)d_in[6];
  const float* b_in  = (const float*)d_in[7];
  const float* g_in  = (const float*)d_in[8];
  const float* be_in = (const float*)d_in[9];
  const float* W_fe  = (const float*)d_in[10];
  const float* b_fe  = (const float*)d_in[11];
  const float* g_fe  = (const float*)d_in[12];
  const float* be_fe = (const float*)d_in[13];
  const float* E_wl  = (const float*)d_in[14];
  const float* E_pos = (const float*)d_in[15];
  const float* E_hop = (const float*)d_in[16];
  const float* W_np  = (const float*)d_in[17];
  const float* b_np  = (const float*)d_in[18];
  const float* Wqkv  = (const float*)d_in[19];
  const float* bqkv  = (const float*)d_in[20];
  const float* Wo    = (const float*)d_in[21];
  const float* bo    = (const float*)d_in[22];
  const float* ln1g  = (const float*)d_in[23];
  const float* ln1b  = (const float*)d_in[24];
  const float* W1    = (const float*)d_in[25];
  const float* b1    = (const float*)d_in[26];
  const float* W2    = (const float*)d_in[27];
  const float* b2    = (const float*)d_in[28];
  const float* ln2g  = (const float*)d_in[29];
  const float* ln2b  = (const float*)d_in[30];
  int n = in_sizes[1];

  char* ws = (char*)d_ws;
  int* starts = (int*)ws;
  int* counts = (int*)(ws + 1024);
  size_t PS = (size_t)NG * SMAX * HD;
  size_t off = 4096;
  float*    p     = (float*)(ws + off);     off += PS * 4;
  _Float16* phalf = (_Float16*)(ws + off);  off += PS * 2;
  _Float16* qkv   = (_Float16*)(ws + off);  off += PS * 3 * 2;
  _Float16* attno = (_Float16*)(ws + off);  off += PS * 2;
  _Float16* W_in16 = (_Float16*)(ws + off); off += (size_t)128 * HD * 2;
  _Float16* W_fe16 = (_Float16*)(ws + off); off += (size_t)HD * HD * 2;
  _Float16* W_np16 = (_Float16*)(ws + off); off += (size_t)1024 * HD * 2;
  _Float16* Wqkv16 = (_Float16*)(ws + off); off += (size_t)LAYERS * HD * 768 * 2;
  _Float16* Wo16   = (_Float16*)(ws + off); off += (size_t)LAYERS * HD * HD * 2;
  _Float16* W116   = (_Float16*)(ws + off); off += (size_t)LAYERS * HD * 1024 * 2;
  _Float16* W216   = (_Float16*)(ws + off); off += (size_t)LAYERS * 1024 * HD * 2;

  // one-time (per call) f16 weight conversion
  k_cvt<<<256, 256, 0, stream>>>(W_in, W_in16, (size_t)128 * HD);
  k_cvt<<<256, 256, 0, stream>>>(W_fe, W_fe16, (size_t)HD * HD);
  k_cvt<<<256, 256, 0, stream>>>(W_np, W_np16, (size_t)1024 * HD);
  k_cvt<<<1024, 256, 0, stream>>>(Wqkv, Wqkv16, (size_t)LAYERS * HD * 768);
  k_cvt<<<1024, 256, 0, stream>>>(Wo, Wo16, (size_t)LAYERS * HD * HD);
  k_cvt<<<1024, 256, 0, stream>>>(W1, W116, (size_t)LAYERS * HD * 1024);
  k_cvt<<<1024, 256, 0, stream>>>(W2, W216, (size_t)LAYERS * 1024 * HD);

  k_starts<<<1, 256, 0, stream>>>(batch, n, NG, counts, starts);
  k_zero<<<4096, 256, 0, stream>>>(p, phalf, PS);

  int tiles = (n + 15) / 16;
  k_embed<<<tiles, 512, 0, stream>>>(x, wl, pos, hop, batch, starts, n,
                                     W_in16, b_in, g_in, be_in,
                                     W_fe16, b_fe, g_fe, be_fe,
                                     E_wl, E_pos, E_hop, W_np16, b_np, p, phalf);

  int rowTiles = NG * SMAX / 16;
  for (int l = 0; l < LAYERS; ++l) {
    k_qkv<<<dim3(NG * SMAX / 64, 6), 256, 0, stream>>>(
        phalf, Wqkv16 + (size_t)l * HD * 768, bqkv + (size_t)l * 3 * HD, qkv);
    k_attn<<<dim3(SMAX / 16, NG), 256, 0, stream>>>(qkv, counts, attno);
    k_otail<<<rowTiles, 512, 0, stream>>>(
        attno, Wo16 + (size_t)l * HD * HD, bo + (size_t)l * HD,
        ln1g + (size_t)l * HD, ln1b + (size_t)l * HD,
        W116 + (size_t)l * HD * 1024, b1 + (size_t)l * 4 * HD,
        W216 + (size_t)l * 1024 * HD, b2 + (size_t)l * HD,
        ln2g + (size_t)l * HD, ln2b + (size_t)l * HD, p, phalf);
  }
  k_gather<<<4096, 256, 0, stream>>>(p, batch, starts, n, (float*)d_out);
}
